// RelPosSelfAttention_72748156060254
// MI455X (gfx1250) — matrix operand order repaired and checked by reading, not compiled, not run on hardware
//
#include <hip/hip_runtime.h>
#include <hip/hip_bf16.h>

#define D_MODEL 1024
#define NHEAD   16
#define HEAD_DIM 64
#define MAX_REL 512
#define BATCH   2
#define SEQ     2048
#define M_TOK   (BATCH * SEQ)   // 4096

typedef __attribute__((ext_vector_type(16))) __bf16 v16bf;
typedef __attribute__((ext_vector_type(8)))  float  v8f;

union Frag16 { v16bf v; unsigned short u[16]; unsigned int d[8]; };

__device__ __forceinline__ unsigned short f2bf(float f) {
  union { float f; unsigned int u; } x; x.f = f;
  unsigned int r = x.u + 0x7fffu + ((x.u >> 16) & 1u);   // round-to-nearest-even
  return (unsigned short)(r >> 16);
}

__device__ __forceinline__ v8f vzero() {
  v8f z;
#pragma unroll
  for (int i = 0; i < 8; ++i) z[i] = 0.0f;
  return z;
}

// ---- WMMA fragment loaders (contiguous k = 16h + s order for A and B, wave32) ----
// A 16x32 bf16 (MxK): lane = M; lanes 0-15 hold K 0..15,
// lanes 16-31 hold K 16..31; loaded as 2x b128.
__device__ __forceinline__ v16bf load_a_frag(const unsigned short* base, int ld, int k0) {
  int lane = threadIdx.x & 31;
  int row = lane & 15, hi = lane >> 4;
  Frag16 f;
  const unsigned int* s0 = (const unsigned int*)(base + (size_t)row * ld + k0 + hi * 16);
  const unsigned int* s1 = (const unsigned int*)(base + (size_t)row * ld + k0 + 8 + hi * 16);
#pragma unroll
  for (int p = 0; p < 4; ++p) { f.d[p] = s0[p]; f.d[4 + p] = s1[p]; }
  return f.v;
}

// B 32x16 bf16 (KxN) sourced from a row-major [N][K] array (Bt):
// lane = N, lanes 0-15 hold K k0..k0+15, lanes 16-31 hold k0+16..k0+31.
__device__ __forceinline__ v16bf load_bt_frag(const unsigned short* bt, int ld, int k0) {
  int lane = threadIdx.x & 31;
  int n = lane & 15, hi = lane >> 4;
  Frag16 f;
  const unsigned int* s = (const unsigned int*)(bt + (size_t)n * ld + k0 + hi * 16);
#pragma unroll
  for (int p = 0; p < 8; ++p) f.d[p] = s[p];
  return f.v;
}

__device__ __forceinline__ v8f wmma_bf16(v16bf a, v16bf b, v8f c) {
  return __builtin_amdgcn_wmma_f32_16x16x32_bf16(false, a, false, b, (short)0, c, false, false);
}

// ---------------- fp32 -> bf16 conversion ----------------
__global__ void cvt_f32_bf16(const float* __restrict__ in, unsigned short* __restrict__ out, int n) {
  int i = blockIdx.x * blockDim.x + threadIdx.x;
  int stride = gridDim.x * blockDim.x;
  for (; i < n; i += stride) out[i] = f2bf(in[i]);
}

// ---------------- GEMM: C[16x64 tile] = A[M,K] * Bt[N,K]^T + bias ----------------
// MODE 0: QKV projection, scatter to Q (scaled 1/8), K, V-transposed bf16 buffers.
// MODE 1: output projection, write fp32 d_out.
template <int MODE>
__global__ void gemm16x64(const unsigned short* __restrict__ A, const unsigned short* __restrict__ Bt,
                          const float* __restrict__ bias, int K, int Ngroups,
                          unsigned short* __restrict__ qb, unsigned short* __restrict__ kb,
                          unsigned short* __restrict__ vt, float* __restrict__ outf) {
  int wave = threadIdx.x >> 5;
  int gid = blockIdx.x * (blockDim.x >> 5) + wave;
  int mt = gid % (M_TOK / 16);
  int ng = gid / (M_TOK / 16);
  if (ng >= Ngroups) return;  // wave-uniform

  int lane = threadIdx.x & 31;
  int hi = lane >> 4, ncol = lane & 15;
  const unsigned short* Abase = A + (size_t)mt * 16 * K;
  int n0 = ng * 64;

  v8f acc[4] = {vzero(), vzero(), vzero(), vzero()};
  for (int k0 = 0; k0 < K; k0 += 32) {
    v16bf a = load_a_frag(Abase, K, k0);
#pragma unroll
    for (int t = 0; t < 4; ++t) {
      v16bf b = load_bt_frag(Bt + (size_t)(n0 + t * 16) * K, K, k0);
      acc[t] = wmma_bf16(a, b, acc[t]);
    }
  }

#pragma unroll
  for (int t = 0; t < 4; ++t) {
#pragma unroll
    for (int r = 0; r < 8; ++r) {
      int m = mt * 16 + r + hi * 8;          // token index
      int n = n0 + t * 16 + ncol;            // output feature
      float c = acc[t][r] + bias[n];
      if (MODE == 0) {
        int chunk = n >> 10, rem = n & 1023;
        int h = rem >> 6, d = rem & 63;
        int bb = m >> 11, ll = m & 2047;
        size_t hd = ((size_t)(bb * NHEAD + h) * SEQ + ll) * HEAD_DIM + d;
        if (chunk == 0)      qb[hd] = f2bf(c * 0.125f);   // fold Dh^-0.5 into Q
        else if (chunk == 1) kb[hd] = f2bf(c);
        else vt[((size_t)(bb * NHEAD + h) * HEAD_DIM + d) * SEQ + ll] = f2bf(c);
      } else {
        outf[(size_t)m * D_MODEL + n] = c;
      }
    }
  }
}

// ---------------- Flash attention with relative-position bias ----------------
// Block = 8 waves, all on the same (batch, head); 32-key chunks of K and
// transposed-V are staged once per block into LDS with async global->LDS
// loads (ASYNCcnt); the per-head bias table (1025 floats) is staged into LDS
// once, so inner-loop bias lookups are ds_load_b32.
__global__ void __launch_bounds__(256, 1)
attn_flash(const unsigned short* __restrict__ Qb, const unsigned short* __restrict__ Kb,
           const unsigned short* __restrict__ Vt, const float* __restrict__ rel_emb,
           unsigned short* __restrict__ attn_out) {
  __shared__ __align__(16) unsigned short kstage[32 * 64];    // [k][d]  4 KB
  __shared__ __align__(16) unsigned short vstage[64 * 32];    // [d][k]  4 KB
  __shared__ __align__(16) unsigned short pbuf[8][16 * 32];   // per-wave P tile
  __shared__ __align__(16) float biash[2 * MAX_REL + 1];      // per-head bias row, 4.1 KB

  int wave = threadIdx.x >> 5, lane = threadIdx.x & 31;
  int t = threadIdx.x;
  int hi = lane >> 4, ncol = lane & 15;

  // block -> (b, h, group of 8 consecutive q-tiles); wave -> one 16-query tile
  int qgrp = blockIdx.x & 15;
  int h    = (blockIdx.x >> 4) & 15;
  int b    = blockIdx.x >> 8;
  int qbase = (qgrp * 8 + wave) * 16;

  // stage this head's bias column once
  for (int i = t; i < 2 * MAX_REL + 1; i += 256) biash[i] = rel_emb[i * NHEAD + h];

  const unsigned short* Qh = Qb + (size_t)(b * NHEAD + h) * SEQ * HEAD_DIM;
  const unsigned short* Kh = Kb + (size_t)(b * NHEAD + h) * SEQ * HEAD_DIM;
  const unsigned short* Vh = Vt + (size_t)(b * NHEAD + h) * HEAD_DIM * SEQ;

  v16bf qa0 = load_a_frag(Qh + (size_t)qbase * HEAD_DIM, HEAD_DIM, 0);
  v16bf qa1 = load_a_frag(Qh + (size_t)qbase * HEAD_DIM, HEAD_DIM, 32);

  float mrow[8], lrow[8];
#pragma unroll
  for (int r = 0; r < 8; ++r) { mrow[r] = -1e30f; lrow[r] = 0.0f; }
  v8f o[4] = {vzero(), vzero(), vzero(), vzero()};

  // per-thread staging addresses (16 B per thread per buffer)
  unsigned kdst = (unsigned)(size_t)&kstage[t * 8];
  unsigned vdst = (unsigned)(size_t)&vstage[(t >> 2) * 32 + (t & 3) * 8];

  for (int k0 = 0; k0 < SEQ; k0 += 32) {
    __syncthreads();  // previous chunk consumed (and, first iter, biash ready)
    const unsigned short* ksrc = Kh + (size_t)k0 * HEAD_DIM + t * 8;            // contiguous 4KB chunk
    const unsigned short* vsrc = Vh + (size_t)(t >> 2) * SEQ + k0 + (t & 3) * 8; // 64B rows of V^T
    asm volatile("global_load_async_to_lds_b128 %0, %1, off" :: "v"(kdst), "v"(ksrc) : "memory");
    asm volatile("global_load_async_to_lds_b128 %0, %1, off" :: "v"(vdst), "v"(vsrc) : "memory");
    asm volatile("s_wait_asynccnt 0x0" ::: "memory");  // own async writes landed in LDS
    __syncthreads();                                    // all waves' stages visible

    // S = Q * K^T  (two 16x16 C tiles, Dh=64 -> 2 K-steps each), K frags from LDS
    v16bf kf = load_bt_frag(&kstage[0], HEAD_DIM, 0);
    v8f s0 = wmma_bf16(qa0, kf, vzero());
    kf = load_bt_frag(&kstage[0], HEAD_DIM, 32);
    s0 = wmma_bf16(qa1, kf, s0);
    kf = load_bt_frag(&kstage[16 * HEAD_DIM], HEAD_DIM, 0);
    v8f s1 = wmma_bf16(qa0, kf, vzero());
    kf = load_bt_frag(&kstage[16 * HEAD_DIM], HEAD_DIM, 32);
    s1 = wmma_bf16(qa1, kf, s1);

    // bias + online softmax (rows live across the 16 lanes of one half)
#pragma unroll
    for (int r = 0; r < 8; ++r) {
      int qpos = qbase + r + hi * 8;
      // symmetric-ternary form -> single v_med3_i32 per clamp
      int rel0 = k0 + ncol - qpos;
      int rel1 = k0 + 16 + ncol - qpos;
      rel0 = rel0 < -MAX_REL ? -MAX_REL : (rel0 > MAX_REL ? MAX_REL : rel0);
      rel1 = rel1 < -MAX_REL ? -MAX_REL : (rel1 > MAX_REL ? MAX_REL : rel1);
      float sv0 = s0[r] + biash[rel0 + MAX_REL];
      float sv1 = s1[r] + biash[rel1 + MAX_REL];
      float rm = fmaxf(sv0, sv1);
#pragma unroll
      for (int off = 8; off >= 1; off >>= 1) rm = fmaxf(rm, __shfl_xor(rm, off, 32));
      float mnew = fmaxf(mrow[r], rm);
      float alpha = __expf(mrow[r] - mnew);
      mrow[r] = mnew;
      float p0 = __expf(sv0 - mnew);
      float p1 = __expf(sv1 - mnew);
      float rs = p0 + p1;
#pragma unroll
      for (int off = 8; off >= 1; off >>= 1) rs += __shfl_xor(rs, off, 32);
      lrow[r] = lrow[r] * alpha + rs;
#pragma unroll
      for (int tt = 0; tt < 4; ++tt) o[tt][r] *= alpha;
      int prow = r + hi * 8;
      pbuf[wave][prow * 32 + ncol]      = f2bf(p0);
      pbuf[wave][prow * 32 + 16 + ncol] = f2bf(p1);
    }
    asm volatile("s_wait_dscnt 0x0" ::: "memory");  // per-wave LDS RAW fence
    v16bf pa = load_a_frag(&pbuf[wave][0], 32, 0);  // C-layout -> A-fragment via LDS
#pragma unroll
    for (int tt = 0; tt < 4; ++tt) {
      v16bf vb = load_bt_frag(&vstage[(tt * 16) * 32], 32, 0);  // V^T frags from LDS
      o[tt] = wmma_bf16(pa, vb, o[tt]);
    }
    asm volatile("" ::: "memory");
  }

  size_t obase = (size_t)b * SEQ * D_MODEL + (size_t)h * HEAD_DIM;
#pragma unroll
  for (int r = 0; r < 8; ++r) {
    float inv = 1.0f / lrow[r];
    int qpos = qbase + r + hi * 8;
#pragma unroll
    for (int tt = 0; tt < 4; ++tt) {
      int d = tt * 16 + ncol;
      attn_out[obase + (size_t)qpos * D_MODEL + d] = f2bf(o[tt][r] * inv);
    }
  }
}

extern "C" void kernel_launch(void* const* d_in, const int* in_sizes, int n_in,
                              void* d_out, int out_size, void* d_ws, size_t ws_size,
                              hipStream_t stream) {
  const float* x       = (const float*)d_in[0];
  const float* qkv_w   = (const float*)d_in[1];
  const float* qkv_b   = (const float*)d_in[2];
  const float* out_w   = (const float*)d_in[3];
  const float* out_b   = (const float*)d_in[4];
  const float* rel_emb = (const float*)d_in[5];
  float* out = (float*)d_out;

  // workspace layout (bytes), total 48 MB
  char* ws = (char*)d_ws;
  unsigned short* xb   = (unsigned short*)(ws + 0);          // 8 MB  x bf16 [4096][1024]
  unsigned short* qwb  = (unsigned short*)(ws + 8388608);    // 6 MB  qkv_w bf16 [3072][1024]
  unsigned short* owb  = (unsigned short*)(ws + 14680064);   // 2 MB  out_w bf16 [1024][1024]
  unsigned short* Qb   = (unsigned short*)(ws + 16777216);   // 8 MB  Q bf16 [b,h,L,64] (pre-scaled)
  unsigned short* Kbuf = (unsigned short*)(ws + 25165824);   // 8 MB  K bf16 [b,h,L,64]
  unsigned short* Vt   = (unsigned short*)(ws + 33554432);   // 8 MB  V bf16 [b,h,64,L]
  unsigned short* attn = (unsigned short*)(ws + 41943040);   // 8 MB  attn out bf16 [4096][1024]

  cvt_f32_bf16<<<1024, 256, 0, stream>>>(x, xb, M_TOK * D_MODEL);
  cvt_f32_bf16<<<1024, 256, 0, stream>>>(qkv_w, qwb, 3 * D_MODEL * D_MODEL);
  cvt_f32_bf16<<<512, 256, 0, stream>>>(out_w, owb, D_MODEL * D_MODEL);

  // QKV: M=4096, N=3072, K=1024 -> 256 M-tiles x 48 N-groups = 12288 waves
  gemm16x64<0><<<1536, 256, 0, stream>>>(xb, qwb, qkv_b, D_MODEL, 48,
                                         Qb, Kbuf, Vt, nullptr);
  // attention: 512 blocks = (b=2) x (h=16) x (16 q-groups), 8 q-tile waves each
  attn_flash<<<512, 256, 0, stream>>>(Qb, Kbuf, Vt, rel_emb, attn);
  // out-proj: M=4096, N=1024, K=1024 -> 256 x 16 = 4096 waves
  gemm16x64<1><<<512, 256, 0, stream>>>(attn, owb, out_b, D_MODEL, 16,
                                        nullptr, nullptr, nullptr, out);
}
